// FlashLinearAttention_35433480192035
// MI455X (gfx1250) — matrix operand order repaired and checked by reading, compiled, not run on hardware
//
#include <hip/hip_runtime.h>

#define BB 8
#define TT 4096
#define DD 512
#define SS 128
#define EE 1024
#define CK 64
#define GG (TT / CK)
#define MTOT (BB * TT)

typedef __attribute__((ext_vector_type(16))) __bf16 v16bf;
typedef __attribute__((ext_vector_type(8))) float v8f;

struct __attribute__((aligned(16))) bf16x8 {
  __bf16 h[8];
};

__device__ inline v8f zero8() {
  v8f z = {0.f, 0.f, 0.f, 0.f, 0.f, 0.f, 0.f, 0.f};
  return z;
}

__device__ inline v8f wmma_bf16(v16bf a, v16bf b, v8f c) {
  return __builtin_amdgcn_wmma_f32_16x16x32_bf16(false, a, false, b, (short)0, c,
                                                 false, false);
}

// A fragment: 16x32 bf16 from row-major A[M x K] (lda = K).
// Lane L (grp = L>>4): row = m0+(L&15); elems 0..15 = K[k0+16*grp .. +15].
__device__ inline v16bf load_frag_a(const __bf16* A, int lda, int m0, int k0) {
  int lane = threadIdx.x & 31;
  const __bf16* p = A + (size_t)(m0 + (lane & 15)) * lda + k0 + ((lane >> 4) << 4);
  v16bf r;
#pragma unroll
  for (int i = 0; i < 8; ++i) r[i] = p[i];
#pragma unroll
  for (int i = 0; i < 8; ++i) r[8 + i] = p[8 + i];
  return r;
}

// B fragment: 32x16 where B[k][n] = W[n][k], W row-major [N x K] (ldw = K).
// Lane L (grp): col = n0+(L&15); elems = K[k0+16*grp .. +15].
__device__ inline v16bf load_frag_b(const __bf16* W, int ldw, int n0, int k0) {
  int lane = threadIdx.x & 31;
  const __bf16* p = W + (size_t)(n0 + (lane & 15)) * ldw + k0 + ((lane >> 4) << 4);
  v16bf r;
#pragma unroll
  for (int i = 0; i < 16; ++i) r[i] = p[i];
  return r;
}

// ---------------- elementwise helpers ----------------

__global__ __launch_bounds__(256) void f2bf_kernel(const float* __restrict__ in,
                                                   __bf16* __restrict__ out, int n) {
  int i = blockIdx.x * 256 + threadIdx.x;
  if (i < n) out[i] = (__bf16)in[i];
}

// ScaleNorm: x = q * rsqrt(mean(q^2)+eps) * scala   -> bf16
__global__ __launch_bounds__(256) void scalenorm_kernel(const float* __restrict__ q,
                                                        const float* __restrict__ scala,
                                                        __bf16* __restrict__ xout) {
  int wave = threadIdx.x >> 5, lane = threadIdx.x & 31;
  size_t row = (size_t)blockIdx.x * 8 + wave;
  const float* p = q + row * DD;
  float ss = 0.f;
#pragma unroll
  for (int i = 0; i < DD / 32; ++i) {
    float val = p[lane + i * 32];
    ss += val * val;
  }
#pragma unroll
  for (int off = 16; off > 0; off >>= 1) ss += __shfl_xor(ss, off, 32);
  float scale = rsqrtf(ss * (1.f / DD) + 1e-5f) * scala[0];
  __bf16* o = xout + row * DD;
#pragma unroll
  for (int i = 0; i < DD / 32; ++i) o[lane + i * 32] = (__bf16)(p[lane + i * 32] * scale);
}

// ---------------- generic WMMA GEMM:  C = act(A @ W^T + bias) ----------------
// A bf16 [M,K] row-major, W bf16 [N,K] row-major. Wave tile 64x64, block 8 waves
// stacked over M -> block tile 512x64. The 64x32 weight tile is staged into LDS
// once per block per k-step via async global->LDS DMA (ASYNCcnt path), so the
// 8 waves share it instead of re-reading global memory.
template <bool SILU, bool OUTF, bool OUTB>
__global__ __launch_bounds__(256) void gemm_bias_act(
    const __bf16* __restrict__ A, const __bf16* __restrict__ W,
    const float* __restrict__ bias, float* __restrict__ outF,
    __bf16* __restrict__ outB, int M, int N, int K) {
  __shared__ __bf16 wtile[64][32];  // [n_local][k_local], 4KB
  int wave = threadIdx.x >> 5;
  int lane = threadIdx.x & 31;
  int m0 = blockIdx.x * 512 + wave * 64;
  int n0 = blockIdx.y * 64;
  int sn = threadIdx.x >> 2;        // 0..63  staging row (N)
  int sk = (threadIdx.x & 3) * 8;   // 0,8,16,24 staging K offset
  unsigned ldsoff = (unsigned)(size_t)&wtile[sn][sk];

  v8f acc[4][4];
#pragma unroll
  for (int i = 0; i < 4; ++i)
#pragma unroll
    for (int j = 0; j < 4; ++j) acc[i][j] = zero8();

  for (int k0 = 0; k0 < K; k0 += 32) {
    // async stage: 256 threads x 16B = the whole 64x32 bf16 tile
    const __bf16* gp = W + (size_t)(n0 + sn) * K + k0 + sk;
    asm volatile("global_load_async_to_lds_b128 %0, %1, off"
                 :
                 : "v"(ldsoff), "v"(gp)
                 : "memory");
    if (k0 + 32 < K)
      __builtin_prefetch(A + (size_t)(m0 + (lane & 15)) * K + k0 + 32, 0, 1);
    asm volatile("s_wait_asynccnt 0x0" : : : "memory");
    __syncthreads();

    v16bf bfrag[4];
#pragma unroll
    for (int j = 0; j < 4; ++j) bfrag[j] = load_frag_b(&wtile[0][0], 32, j * 16, 0);
#pragma unroll
    for (int i = 0; i < 4; ++i) {
      v16bf afrag = load_frag_a(A, K, m0 + i * 16, k0);
#pragma unroll
      for (int j = 0; j < 4; ++j) acc[i][j] = wmma_bf16(afrag, bfrag[j], acc[i][j]);
    }
    __syncthreads();  // protect wtile before next-iteration staging
  }
  int nl = lane & 15, mg = (lane >> 4) * 8;
#pragma unroll
  for (int i = 0; i < 4; ++i)
#pragma unroll
    for (int j = 0; j < 4; ++j) {
      int col = n0 + j * 16 + nl;
      float bcol = bias[col];
#pragma unroll
      for (int r = 0; r < 8; ++r) {
        size_t row = (size_t)(m0 + i * 16 + mg + r);
        float val = acc[i][j][r] + bcol;
        if (SILU) val = val / (1.f + __expf(-val));
        if (OUTF) outF[row * N + col] = val;
        if (OUTB) outB[row * N + col] = (__bf16)val;
      }
    }
}

// ---------------- heads + RoPE ----------------

__device__ inline void rope_emit(__bf16* __restrict__ out, const float* __restrict__ wt,
                                 const float* __restrict__ bs, size_t row, int i,
                                 float x1, float x2, float sn, float cs) {
  float h1 = x1 * wt[i] + bs[i];
  float h2 = x2 * wt[i + 64] + bs[i + 64];
  out[row * SS + i] = (__bf16)(h1 * cs - h2 * sn);
  out[row * SS + 64 + i] = (__bf16)(h2 * cs + h1 * sn);
}

__global__ __launch_bounds__(256) void heads_rope_kernel(
    const float* __restrict__ base, const float* __restrict__ qqw,
    const float* __restrict__ qqb, const float* __restrict__ qkw,
    const float* __restrict__ qkb, const float* __restrict__ lqw,
    const float* __restrict__ lqb, const float* __restrict__ lkw,
    const float* __restrict__ lkb, __bf16* __restrict__ qq, __bf16* __restrict__ qk,
    __bf16* __restrict__ lq, __bf16* __restrict__ lk) {
  int idx = blockIdx.x * 256 + threadIdx.x;
  size_t row = (size_t)(idx >> 6);
  int i = idx & 63;
  int t = (int)(row & (TT - 1));  // position within batch = g*CK + n
  float x1 = base[row * SS + i];
  float x2 = base[row * SS + 64 + i];
  float ang = (float)t * __powf(10000.f, -(float)i * (1.f / 64.f));
  float sn, cs;
  __sincosf(ang, &sn, &cs);
  rope_emit(qq, qqw, qqb, row, i, x1, x2, sn, cs);
  rope_emit(qk, qkw, qkb, row, i, x1, x2, sn, cs);
  rope_emit(lq, lqw, lqb, row, i, x1, x2, sn, cs);
  rope_emit(lk, lkw, lkb, row, i, x1, x2, sn, cs);
}

// ---------------- lin_kv = (1/CK) * sum_g lin_k^T @ v_c, stored transposed [B,E,S] ----

__global__ __launch_bounds__(256) void linkv_kernel(const __bf16* __restrict__ lin_k,
                                                    const __bf16* __restrict__ v,
                                                    __bf16* __restrict__ linkvT) {
  __shared__ __bf16 kT[SS][CK];  // [s][n]  (A = lin_k^T, row-major M=s, K=n)
  __shared__ __bf16 vT[64][CK];  // [e_local][n]  (B = v_c, col-contiguous)
  int b = blockIdx.x;
  int e0 = blockIdx.y * 64;
  int wave = threadIdx.x >> 5, lane = threadIdx.x & 31;
  int tn = threadIdx.x >> 2;  // 0..63 staging chunk-row n
  v8f acc[4];
#pragma unroll
  for (int j = 0; j < 4; ++j) acc[j] = zero8();

  for (int g = 0; g < GG; ++g) {
    size_t rowbase = (size_t)b * TT + (size_t)g * CK;
    // stage lin_k chunk transposed: vectorized 16B global loads, scatter ds stores
    {
      int sb = (threadIdx.x & 3) * 32;
      const __bf16* kp = lin_k + (rowbase + tn) * SS + sb;
#pragma unroll
      for (int c = 0; c < 4; ++c) {
        bf16x8 ch = *(const bf16x8*)(kp + c * 8);
#pragma unroll
        for (int e = 0; e < 8; ++e) kT[sb + c * 8 + e][tn] = ch.h[e];
      }
    }
    {
      int eb = (threadIdx.x & 3) * 16;
      const __bf16* vp = v + (rowbase + tn) * EE + e0 + eb;
#pragma unroll
      for (int c = 0; c < 2; ++c) {
        bf16x8 ch = *(const bf16x8*)(vp + c * 8);
#pragma unroll
        for (int e = 0; e < 8; ++e) vT[eb + c * 8 + e][tn] = ch.h[e];
      }
    }
    __syncthreads();
    int m0 = wave * 16;  // 8 waves cover S=128
#pragma unroll
    for (int kk = 0; kk < 2; ++kk) {
      v16bf afrag = load_frag_a(&kT[0][0], CK, m0, kk * 32);
#pragma unroll
      for (int j = 0; j < 4; ++j) {
        v16bf bfrag = load_frag_b(&vT[0][0], CK, j * 16, kk * 32);
        acc[j] = wmma_bf16(afrag, bfrag, acc[j]);
      }
    }
    __syncthreads();
  }
  int nl = lane & 15, mg = (lane >> 4) * 8;
#pragma unroll
  for (int j = 0; j < 4; ++j) {
    int e = e0 + j * 16 + nl;
#pragma unroll
    for (int r = 0; r < 8; ++r) {
      int s = wave * 16 + mg + r;
      linkvT[((size_t)b * EE + e) * SS + s] = (__bf16)(acc[j][r] * (1.f / CK));
    }
  }
}

// ---------------- per-chunk: kern = relu(qk/CK + bias)^2; out = u*(kern@v + lin_q@lin_kv)

__global__ __launch_bounds__(256) void quad_lin_combine(
    const __bf16* __restrict__ quad_q, const __bf16* __restrict__ quad_k,
    const __bf16* __restrict__ lin_q, const __bf16* __restrict__ v,
    const __bf16* __restrict__ u, const __bf16* __restrict__ linkvT,
    const float* __restrict__ wrel, __bf16* __restrict__ combined) {
  __shared__ __bf16 kern[CK][CK];  // [n][m] bf16 (A of second GEMM)
  __shared__ __bf16 vT[128][CK];   // [e_local][m] transposed v slice (B operand)
  int b = blockIdx.x >> 6, g = blockIdx.x & 63;
  size_t rowbase = (size_t)b * TT + (size_t)g * CK;
  int wave = threadIdx.x >> 5, lane = threadIdx.x & 31;
  int nl = lane & 15, mg = (lane >> 4) * 8;
  const __bf16* Q = quad_q + rowbase * SS;
  const __bf16* Km = quad_k + rowbase * SS;

  // phase 1: qk (64x64, K=128) -> kern in LDS. 8 waves x 2 tiles = 16 tiles.
  {
    int mi = wave >> 1;
#pragma unroll
    for (int jj = 0; jj < 2; ++jj) {
      int nj = (wave & 1) * 2 + jj;
      v8f acc = zero8();
#pragma unroll
      for (int kk = 0; kk < 4; ++kk)
        acc = wmma_bf16(load_frag_a(Q, SS, mi * 16, kk * 32),
                        load_frag_b(Km, SS, nj * 16, kk * 32), acc);
#pragma unroll
      for (int r = 0; r < 8; ++r) {
        int nrow = mi * 16 + mg + r;
        int mcol = nj * 16 + nl;
        float val = acc[r] * (1.f / CK) + wrel[511 + mcol - nrow];  // Toeplitz bias
        val = fmaxf(val, 0.f);
        kern[nrow][mcol] = (__bf16)(val * val);
      }
    }
  }
  __syncthreads();

  const __bf16* LQ = lin_q + rowbase * SS;
  const __bf16* KV = linkvT + (size_t)b * EE * SS;  // [E][S]
  int i = wave >> 1;
  int jbase = (wave & 1) * 4;
  int tr = threadIdx.x >> 2;         // 0..63 staging chunk-row m
  int cb = (threadIdx.x & 3) * 32;   // staging col base within 128
  for (int ec = 0; ec < 8; ++ec) {   // E in chunks of 128 columns
    int e0 = ec * 128;
    {
      const __bf16* vp = v + (rowbase + tr) * EE + e0 + cb;
#pragma unroll
      for (int c = 0; c < 4; ++c) {
        bf16x8 ch = *(const bf16x8*)(vp + c * 8);
#pragma unroll
        for (int e = 0; e < 8; ++e) vT[cb + c * 8 + e][tr] = ch.h[e];
      }
    }
    __syncthreads();
#pragma unroll
    for (int jo = 0; jo < 4; ++jo) {
      int j = jbase + jo;
      int eg = e0 + j * 16;
      v8f acc = zero8();
#pragma unroll
      for (int kk = 0; kk < 2; ++kk)  // quadratic part, K=64
        acc = wmma_bf16(load_frag_a(&kern[0][0], CK, i * 16, kk * 32),
                        load_frag_b(&vT[0][0], CK, j * 16, kk * 32), acc);
#pragma unroll
      for (int kk = 0; kk < 4; ++kk)  // linear part, K=128
        acc = wmma_bf16(load_frag_a(LQ, SS, i * 16, kk * 32),
                        load_frag_b(KV, SS, eg, kk * 32), acc);
#pragma unroll
      for (int r = 0; r < 8; ++r) {
        size_t row = rowbase + i * 16 + mg + r;
        int e = eg + nl;
        float uu = (float)u[row * EE + e];
        combined[row * EE + e] = (__bf16)(uu * acc[r]);
      }
    }
    __syncthreads();
  }
}

// ---------------- launch ----------------

extern "C" void kernel_launch(void* const* d_in, const int* in_sizes, int n_in,
                              void* d_out, int out_size, void* d_ws, size_t ws_size,
                              hipStream_t stream) {
  (void)in_sizes; (void)n_in; (void)out_size; (void)ws_size;
  const float* query = (const float*)d_in[0];
  const float* scala = (const float*)d_in[1];
  const float* u_w = (const float*)d_in[2];
  const float* u_b = (const float*)d_in[3];
  const float* v_w = (const float*)d_in[4];
  const float* v_b = (const float*)d_in[5];
  const float* base_w = (const float*)d_in[6];
  const float* base_b = (const float*)d_in[7];
  const float* qq_w = (const float*)d_in[8];
  const float* qq_b = (const float*)d_in[9];
  const float* qk_w = (const float*)d_in[10];
  const float* qk_b = (const float*)d_in[11];
  const float* lq_w = (const float*)d_in[12];
  const float* lq_b = (const float*)d_in[13];
  const float* lk_w = (const float*)d_in[14];
  const float* lk_b = (const float*)d_in[15];
  const float* o_w = (const float*)d_in[16];
  const float* o_b = (const float*)d_in[17];
  const float* wrel = (const float*)d_in[18];
  float* out = (float*)d_out;

  char* cur = (char*)d_ws;
  auto alloc = [&](size_t bytes) -> char* {
    char* p = cur;
    cur += (bytes + 255) & ~(size_t)255;
    return p;
  };
  __bf16* x_bf = (__bf16*)alloc((size_t)MTOT * DD * 2);
  __bf16* uw_bf = (__bf16*)alloc((size_t)EE * DD * 2);
  __bf16* vw_bf = (__bf16*)alloc((size_t)EE * DD * 2);
  __bf16* bw_bf = (__bf16*)alloc((size_t)SS * DD * 2);
  __bf16* ow_bf = (__bf16*)alloc((size_t)DD * EE * 2);
  __bf16* u_bf = (__bf16*)alloc((size_t)MTOT * EE * 2);
  __bf16* v_bf = (__bf16*)alloc((size_t)MTOT * EE * 2);
  float* base_f = (float*)alloc((size_t)MTOT * SS * 4);
  __bf16* qq_bf = (__bf16*)alloc((size_t)MTOT * SS * 2);
  __bf16* qk_bf = (__bf16*)alloc((size_t)MTOT * SS * 2);
  __bf16* lq_bf = (__bf16*)alloc((size_t)MTOT * SS * 2);
  __bf16* lk_bf = (__bf16*)alloc((size_t)MTOT * SS * 2);
  __bf16* linkvT = (__bf16*)alloc((size_t)BB * EE * SS * 2);
  __bf16* comb_bf = (__bf16*)alloc((size_t)MTOT * EE * 2);

  dim3 blk(256);
  f2bf_kernel<<<(EE * DD + 255) / 256, blk, 0, stream>>>(u_w, uw_bf, EE * DD);
  f2bf_kernel<<<(EE * DD + 255) / 256, blk, 0, stream>>>(v_w, vw_bf, EE * DD);
  f2bf_kernel<<<(SS * DD + 255) / 256, blk, 0, stream>>>(base_w, bw_bf, SS * DD);
  f2bf_kernel<<<(DD * EE + 255) / 256, blk, 0, stream>>>(o_w, ow_bf, DD * EE);

  scalenorm_kernel<<<MTOT / 8, blk, 0, stream>>>(query, scala, x_bf);

  gemm_bias_act<true, false, true><<<dim3(MTOT / 512, EE / 64), blk, 0, stream>>>(
      x_bf, uw_bf, u_b, nullptr, u_bf, MTOT, EE, DD);
  gemm_bias_act<true, false, true><<<dim3(MTOT / 512, EE / 64), blk, 0, stream>>>(
      x_bf, vw_bf, v_b, nullptr, v_bf, MTOT, EE, DD);
  gemm_bias_act<true, true, false><<<dim3(MTOT / 512, SS / 64), blk, 0, stream>>>(
      x_bf, bw_bf, base_b, base_f, nullptr, MTOT, SS, DD);

  heads_rope_kernel<<<(MTOT * 64) / 256, blk, 0, stream>>>(
      base_f, qq_w, qq_b, qk_w, qk_b, lq_w, lq_b, lk_w, lk_b, qq_bf, qk_bf, lq_bf,
      lk_bf);

  linkv_kernel<<<dim3(BB, EE / 64), blk, 0, stream>>>(lk_bf, v_bf, linkvT);

  quad_lin_combine<<<BB * GG, blk, 0, stream>>>(qq_bf, qk_bf, lq_bf, v_bf, u_bf,
                                                linkvT, wrel, comb_bf);

  gemm_bias_act<false, true, false><<<dim3(MTOT / 512, DD / 64), blk, 0, stream>>>(
      comb_bf, ow_bf, o_b, out, nullptr, MTOT, DD, EE);
}
